// RNAModel_13786845020545
// MI455X (gfx1250) — compile-verified, matrix operand order checked by reading, not run on hardware
//
#include <hip/hip_runtime.h>
#include <math.h>

// Model constants (from reference setup_inputs)
#define B_  2
#define L_  256
#define H_  256
#define NH_ 4
#define HD_ 64
#define FF_ 2048
#define C_  128

typedef __bf16 bf16_t;
typedef __attribute__((ext_vector_type(16))) __bf16 v16bf;
typedef __attribute__((ext_vector_type(8)))  float  v8f;
typedef float    f32x4 __attribute__((ext_vector_type(4)));
typedef unsigned u32x4 __attribute__((ext_vector_type(4)));
typedef unsigned u32x2 __attribute__((ext_vector_type(2)));

// ---------------------------------------------------------------------------
// bf16 helpers (round-to-nearest-even; bit-level, no __bf16 arithmetic)
// ---------------------------------------------------------------------------
__device__ inline unsigned short f2bfu(float f) {
  union { float f; unsigned u; } v; v.f = f;
  unsigned r = v.u + 0x7FFFu + ((v.u >> 16) & 1u);
  return (unsigned short)(r >> 16);
}
__device__ inline bf16_t f2bf(float f) {
  unsigned short h = f2bfu(f);
  bf16_t b; __builtin_memcpy(&b, &h, 2); return b;
}
__device__ inline unsigned short bfbits(bf16_t b) {
  unsigned short h; __builtin_memcpy(&h, &b, 2); return h;
}
// pack 4 floats -> 4 bf16 -> single 64-bit store
__device__ inline void store4bf(bf16_t* dst, f32x4 v) {
  u32x2 p;
  p.x = (unsigned)f2bfu(v.x) | ((unsigned)f2bfu(v.y) << 16);
  p.y = (unsigned)f2bfu(v.z) | ((unsigned)f2bfu(v.w) << 16);
  *(u32x2*)dst = p;
}

union Frag { u32x4 q[2]; v16bf v; };

// ---------------------------------------------------------------------------
// Generic batched WMMA GEMM:  C = epi(alpha * A @ op(B) + bias[n])
//   TIN: float (f32->bf16 convert during LDS staging) or bf16_t (native)
//   EPI: 0 none, 1 relu, 2 sigmoid, 3 sigmoid(aux)*acc, 4 acc*aux,
//        5 acc + aux[row] (row-broadcast bias, triangle attention)
//   TRANSB: 1 -> C=A@B^T (B row-major [N,K]); 0 -> C=A@B (B row-major [K,N])
// EXACT TILING REQUIRED: M,N multiples of 64; K multiple of 32 (true for all
// shapes in this model) -> no bounds checks, no exec masking, EXEC==all-1s at
// every v_wmma as the ISA requires.
// ---------------------------------------------------------------------------
struct GemmArgs {
  const void* A; const void* Bm; const float* bias; const float* aux; float* C;
  int M, N, K, lda, ldb, ldc, transB, nb1, nb2;
  float alpha;
  long sA1, sA2, sB1, sB2, sC1, sC2, sX1, sX2;
};

template <typename TIN, int EPI, int TRANSB>
__global__ __launch_bounds__(128) void gemm_wmma(GemmArgs g) {
  constexpr int BM = 64, BN = 64, BK = 32;
  constexpr int AP = BK + 8;  // LDS pitch in halfs: 80B rows keep 16B alignment
  __shared__ bf16_t Asb[2][BM * AP];
  __shared__ bf16_t Bsb[2][BN * AP];   // always n-major [n][k]

  const int tid = threadIdx.x;
  const int bz  = blockIdx.z;
  const int b1  = bz / g.nb2, b2 = bz % g.nb2;
  const TIN*   A   = (const TIN*)g.A  + b1 * g.sA1 + b2 * g.sA2;
  const TIN*   Bm  = (const TIN*)g.Bm + b1 * g.sB1 + b2 * g.sB2;
  float*       Cp  = g.C + b1 * g.sC1 + b2 * g.sC2;
  const float* aux = g.aux ? g.aux + b1 * g.sX1 + b2 * g.sX2 : (const float*)0;

  const int m0 = blockIdx.x * BM, n0 = blockIdx.y * BN;
  const int wv = tid >> 5, lane = tid & 31;
  const int half = lane >> 4, l16 = lane & 15;
  const int wm = (wv >> 1) * 32, wn = (wv & 1) * 32;

  f32x4 aF[4], bF[4];
  u32x4 aH[2], bH[2];

  auto loadTiles = [&](int k0) {
    if constexpr (sizeof(TIN) == 4) {
      const float* Af = (const float*)A;
      const float* Bf = (const float*)Bm;
      const int ar = tid >> 3, ac = (tid & 7) * 4;
#pragma unroll
      for (int it = 0; it < 4; ++it)
        aF[it] = *(const f32x4*)(Af + (long)(m0 + ar + it * 16) * g.lda + (k0 + ac));
      if (TRANSB) {
#pragma unroll
        for (int it = 0; it < 4; ++it)
          bF[it] = *(const f32x4*)(Bf + (long)(n0 + ar + it * 16) * g.ldb + (k0 + ac));
      } else {
        const int br = tid >> 4, bc = (tid & 15) * 4;
#pragma unroll
        for (int it = 0; it < 4; ++it)
          bF[it] = *(const f32x4*)(Bf + (long)(k0 + br + it * 8) * g.ldb + (n0 + bc));
      }
    } else {
      const bf16_t* Ah = (const bf16_t*)A;
      const bf16_t* Bh = (const bf16_t*)Bm;
      const int ar = tid >> 2, ac = (tid & 3) * 8;
#pragma unroll
      for (int it = 0; it < 2; ++it) {
        aH[it] = *(const u32x4*)(Ah + (long)(m0 + ar + it * 32) * g.lda + (k0 + ac));
        bH[it] = *(const u32x4*)(Bh + (long)(n0 + ar + it * 32) * g.ldb + (k0 + ac));
      }
    }
  };

  auto storeTiles = [&](int buf) {
    if constexpr (sizeof(TIN) == 4) {
      const int ar = tid >> 3, ac = (tid & 7) * 4;
#pragma unroll
      for (int it = 0; it < 4; ++it)
        store4bf(&Asb[buf][(ar + it * 16) * AP + ac], aF[it]);
      if (TRANSB) {
#pragma unroll
        for (int it = 0; it < 4; ++it)
          store4bf(&Bsb[buf][(ar + it * 16) * AP + ac], bF[it]);
      } else {
        // transpose-on-store so fragment loads stay vectorized
        const int br = tid >> 4, bc = (tid & 15) * 4;
#pragma unroll
        for (int it = 0; it < 4; ++it) {
          int kk = br + it * 8;
          Bsb[buf][(bc + 0) * AP + kk] = f2bf(bF[it].x);
          Bsb[buf][(bc + 1) * AP + kk] = f2bf(bF[it].y);
          Bsb[buf][(bc + 2) * AP + kk] = f2bf(bF[it].z);
          Bsb[buf][(bc + 3) * AP + kk] = f2bf(bF[it].w);
        }
      }
    } else {
      const int ar = tid >> 2, ac = (tid & 3) * 8;
#pragma unroll
      for (int it = 0; it < 2; ++it) {
        *(u32x4*)&Asb[buf][(ar + it * 32) * AP + ac] = aH[it];
        *(u32x4*)&Bsb[buf][(ar + it * 32) * AP + ac] = bH[it];
      }
    }
  };

  v8f acc[2][2] = {};
  const int KT = g.K / BK;
  loadTiles(0);
  storeTiles(0);
  __syncthreads();
  int cur = 0;
  for (int kt = 0; kt < KT; ++kt) {
    if (kt + 1 < KT) loadTiles((kt + 1) * BK);  // overlap global loads with WMMA

    // CDNA5 wave32 16x16x32 operand layout: lane's 16 halfs are two contiguous
    // 8-half runs at k = half*8 and k = 16 + half*8  -> two ds_load_b128 each.
    Frag fa[2], fb[2];
#pragma unroll
    for (int mi = 0; mi < 2; ++mi) {
      const bf16_t* row = &Asb[cur][(wm + mi * 16 + l16) * AP];
      fa[mi].q[0] = *(const u32x4*)(row + half * 8);
      fa[mi].q[1] = *(const u32x4*)(row + 16 + half * 8);
    }
#pragma unroll
    for (int ni = 0; ni < 2; ++ni) {
      const bf16_t* row = &Bsb[cur][(wn + ni * 16 + l16) * AP];
      fb[ni].q[0] = *(const u32x4*)(row + half * 8);
      fb[ni].q[1] = *(const u32x4*)(row + 16 + half * 8);
    }
#pragma unroll
    for (int mi = 0; mi < 2; ++mi)
#pragma unroll
      for (int ni = 0; ni < 2; ++ni)
        acc[mi][ni] = __builtin_amdgcn_wmma_f32_16x16x32_bf16(
            false, fa[mi].v, false, fb[ni].v, (short)0, acc[mi][ni], false, false);

    if (kt + 1 < KT) {
      storeTiles(cur ^ 1);
      __syncthreads();
      cur ^= 1;
    }
  }

  // Epilogue. C/D layout: vgpr r -> M = r + 8*(lane>=16), N = lane&15.
#pragma unroll
  for (int mi = 0; mi < 2; ++mi)
#pragma unroll
    for (int ni = 0; ni < 2; ++ni) {
      int n = n0 + wn + ni * 16 + l16;
      float bn = g.bias ? g.bias[n] : 0.f;
#pragma unroll
      for (int r = 0; r < 8; ++r) {
        int m = m0 + wm + mi * 16 + half * 8 + r;
        float v = acc[mi][ni][r] * g.alpha + bn;
        long idx = (long)m * g.ldc + n;
        if      (EPI == 1) v = fmaxf(v, 0.f);
        else if (EPI == 2) v = 1.f / (1.f + __expf(-v));
        else if (EPI == 3) v = v / (1.f + __expf(-aux[idx]));   // sigmoid(aux)*acc
        else if (EPI == 4) v = v * aux[idx];
        else if (EPI == 5) v = v + aux[m];
        Cp[idx] = v;
      }
    }
}

// ---------------------------------------------------------------------------
// Wave-per-row LayerNorm (optional residual add), cols in {128,256}, float4 IO
// ---------------------------------------------------------------------------
__global__ __launch_bounds__(256) void ln_kernel(const float* x, const float* res,
    const float* sc, const float* bi, float* out, long rows, int cols) {
  long row = (long)blockIdx.x * 8 + (threadIdx.x >> 5);
  int lane = threadIdx.x & 31;
  if (row >= rows) return;
  const float* xr = x + row * (long)cols;
  const float* rr = res ? res + row * (long)cols : (const float*)0;
  int nv = cols >> 7;  // f32x4 per lane
  f32x4 v4[2];
  float sum = 0.f;
  for (int i = 0; i < nv; ++i) {
    f32x4 v = *(const f32x4*)(xr + lane * 4 + i * 128);
    if (rr) v += *(const f32x4*)(rr + lane * 4 + i * 128);
    v4[i] = v;
    sum += v.x + v.y + v.z + v.w;
  }
  for (int o = 16; o; o >>= 1) sum += __shfl_xor(sum, o, 32);
  float mu = sum / cols;
  float var = 0.f;
  for (int i = 0; i < nv; ++i) {
    f32x4 d = v4[i] - mu;
    var += d.x * d.x + d.y * d.y + d.z * d.z + d.w * d.w;
  }
  for (int o = 16; o; o >>= 1) var += __shfl_xor(var, o, 32);
  float inv = rsqrtf(var / cols + 1e-5f);
  for (int i = 0; i < nv; ++i) {
    int c = lane * 4 + i * 128;
    f32x4 s4 = *(const f32x4*)(sc + c);
    f32x4 b4 = *(const f32x4*)(bi + c);
    f32x4 o4 = (v4[i] - mu) * inv * s4 + b4;
    *(f32x4*)(out + row * (long)cols + c) = o4;
  }
}

// Wave-per-row softmax over last axis (encoder attention), float4 IO
__global__ __launch_bounds__(256) void softmax_rows(float* s, long rows, int cols) {
  long row = (long)blockIdx.x * 8 + (threadIdx.x >> 5);
  int lane = threadIdx.x & 31;
  if (row >= rows) return;
  float* r = s + row * (long)cols;
  int nv = cols >> 7;
  f32x4 v4[2];
  float mx = -1e30f;
  for (int i = 0; i < nv; ++i) {
    v4[i] = *(const f32x4*)(r + lane * 4 + i * 128);
    mx = fmaxf(mx, fmaxf(fmaxf(v4[i].x, v4[i].y), fmaxf(v4[i].z, v4[i].w)));
  }
  for (int o = 16; o; o >>= 1) mx = fmaxf(mx, __shfl_xor(mx, o, 32));
  float sum = 0.f;
  for (int i = 0; i < nv; ++i) {
    v4[i].x = __expf(v4[i].x - mx); v4[i].y = __expf(v4[i].y - mx);
    v4[i].z = __expf(v4[i].z - mx); v4[i].w = __expf(v4[i].w - mx);
    sum += v4[i].x + v4[i].y + v4[i].z + v4[i].w;
  }
  for (int o = 16; o; o >>= 1) sum += __shfl_xor(sum, o, 32);
  float inv = 1.f / sum;
  for (int i = 0; i < nv; ++i)
    *(f32x4*)(r + lane * 4 + i * 128) = v4[i] * inv;
}

// Triangle attention softmax over dim=-2 (j axis of s[b,i,j,k]).
// Block = one (b,i); thread = k; j-loop reads are k-coalesced.
__global__ __launch_bounds__(256) void softmax_dim2(float* s) {
  long base = (long)blockIdx.x * L_ * L_;
  int k = threadIdx.x;
  float mx = -1e30f;
  for (int j = 0; j < L_; ++j) mx = fmaxf(mx, s[base + (long)j * L_ + k]);
  float sum = 0.f;
  for (int j = 0; j < L_; ++j) sum += __expf(s[base + (long)j * L_ + k] - mx);
  float inv = 1.f / sum;
  for (int j = 0; j < L_; ++j) {
    long id = base + (long)j * L_ + k;
    s[id] = __expf(s[id] - mx) * inv;
  }
}

__global__ void embed_kernel(const int* tok, const float* emb, float* x) {
  int row = blockIdx.x, c = threadIdx.x;
  x[(long)row * H_ + c] = emb[(long)tok[row] * H_ + c];
}

// z[b,i,j,c] = pmat[b,i,j]*exp_W[c] + exp_b[c]; 4 rows / block, float4 IO
__global__ __launch_bounds__(128) void expand_kernel(const float* pm, const float* w,
                                                     const float* bv, float* z) {
  long m = (long)blockIdx.x * 4 + (threadIdx.x >> 5);
  int c = (threadIdx.x & 31) * 4;
  float p = pm[m];
  f32x4 wv = *(const f32x4*)(w + c);
  f32x4 bb = *(const f32x4*)(bv + c);
  *(f32x4*)(z + m * C_ + c) = wv * p + bb;
}

// bias[b,i,j] = dot(n[b,i,j,:], Wbias[:,0]); one wave per row, float4 loads
__global__ __launch_bounds__(256) void rowdot_kernel(const float* x, const float* w,
                                                     float* out, long rows) {
  long row = (long)blockIdx.x * 8 + (threadIdx.x >> 5);
  int lane = threadIdx.x & 31;
  if (row >= rows) return;
  f32x4 v  = *(const f32x4*)(x + row * C_ + lane * 4);
  f32x4 w4 = *(const f32x4*)(w + lane * 4);
  float s = v.x * w4.x + v.y * w4.y + v.z * w4.z + v.w * w4.w;
  for (int o = 16; o; o >>= 1) s += __shfl_xor(s, o, 32);
  if (lane == 0) out[row] = s;
}

// a[b,i,k,c] (f32) -> at[(b*C+c), i, k] (bf16); LDS-tiled, 128-bit global IO
__global__ __launch_bounds__(256) void transpose_to_bcik(const float* a, bf16_t* at) {
  __shared__ bf16_t tile[32][36];
  int bi = blockIdx.z, b = bi >> 8, i = bi & 255;
  int k0 = blockIdx.x * 32, c0 = blockIdx.y * 32;
  int tx = threadIdx.x, ty = threadIdx.y;  // (8, 32)
  const float* src = a + (long)bi * L_ * C_;
  f32x4 v = *(const f32x4*)(src + (long)(k0 + ty) * C_ + (c0 + tx * 4));
  store4bf(&tile[ty][tx * 4], v);
  __syncthreads();
  int c = c0 + ty;
  u32x2 p;
  p.x = (unsigned)bfbits(tile[tx * 4 + 0][ty]) | ((unsigned)bfbits(tile[tx * 4 + 1][ty]) << 16);
  p.y = (unsigned)bfbits(tile[tx * 4 + 2][ty]) | ((unsigned)bfbits(tile[tx * 4 + 3][ty]) << 16);
  *(u32x2*)(at + ((long)(b * C_ + c) * L_ + i) * L_ + (k0 + tx * 4)) = p;
}

// ob[(b*C+c), i, j] (f32) -> o2[b,i,j,c] (f32); LDS-tiled, 128-bit global IO
__global__ __launch_bounds__(256) void transpose_to_bijc(const float* ob, float* o2) {
  __shared__ float tile[32][36];
  int bi = blockIdx.z, b = bi >> 8, i = bi & 255;
  int j0 = blockIdx.x * 32, c0 = blockIdx.y * 32;
  int tx = threadIdx.x, ty = threadIdx.y;  // (8, 32)
  f32x4 v = *(const f32x4*)(ob + ((long)(b * C_ + c0 + ty) * L_ + i) * L_ + (j0 + tx * 4));
  *(f32x4*)&tile[ty][tx * 4] = v;
  __syncthreads();
  f32x4 o;
  o.x = tile[tx * 4 + 0][ty]; o.y = tile[tx * 4 + 1][ty];
  o.z = tile[tx * 4 + 2][ty]; o.w = tile[tx * 4 + 3][ty];
  *(f32x4*)(o2 + ((long)bi * L_ + (j0 + ty)) * C_ + (c0 + tx * 4)) = o;
}

// final head: out[m, 0..1] = z[m,:] @ out_W + out_b; one wave per row
__global__ __launch_bounds__(256) void outproj_kernel(const float* z, const float* W,
                                                      const float* bv, float* out, long rows) {
  long row = (long)blockIdx.x * 8 + (threadIdx.x >> 5);
  int lane = threadIdx.x & 31;
  if (row >= rows) return;
  int c = lane * 4;
  f32x4 v  = *(const f32x4*)(z + row * C_ + c);
  f32x4 w0 = *(const f32x4*)(W + c * 2);       // rows c,c+1 of [128,2]
  f32x4 w1 = *(const f32x4*)(W + c * 2 + 4);   // rows c+2,c+3
  float s0 = v.x * w0.x + v.y * w0.z + v.z * w1.x + v.w * w1.z;
  float s1 = v.x * w0.y + v.y * w0.w + v.z * w1.y + v.w * w1.w;
  for (int o = 16; o; o >>= 1) { s0 += __shfl_xor(s0, o, 32); s1 += __shfl_xor(s1, o, 32); }
  if (lane == 0) { out[row * 2] = s0 + bv[0]; out[row * 2 + 1] = s1 + bv[1]; }
}

// ---------------------------------------------------------------------------
// Host-side helpers
// ---------------------------------------------------------------------------
static GemmArgs GA(const void* A, const void* Bm, const float* bias, const float* aux,
                   float* C, int M, int N, int K, int lda, int ldb, int ldc, int transB,
                   float alpha, int nb1 = 1, int nb2 = 1,
                   long sA1 = 0, long sA2 = 0, long sB1 = 0, long sB2 = 0,
                   long sC1 = 0, long sC2 = 0, long sX1 = 0, long sX2 = 0) {
  GemmArgs g;
  g.A = A; g.Bm = Bm; g.bias = bias; g.aux = aux; g.C = C;
  g.M = M; g.N = N; g.K = K; g.lda = lda; g.ldb = ldb; g.ldc = ldc;
  g.transB = transB; g.nb1 = nb1; g.nb2 = nb2; g.alpha = alpha;
  g.sA1 = sA1; g.sA2 = sA2; g.sB1 = sB1; g.sB2 = sB2;
  g.sC1 = sC1; g.sC2 = sC2; g.sX1 = sX1; g.sX2 = sX2;
  return g;
}

static void run_gemm(hipStream_t st, int epi, int bf16in, GemmArgs g) {
  dim3 grid(g.M / 64, g.N / 64, g.nb1 * g.nb2);
  dim3 blk(128, 1, 1);
  if (bf16in) { gemm_wmma<bf16_t, 0, 1><<<grid, blk, 0, st>>>(g); return; }
  if (g.transB) {
    switch (epi) {
      case 5:  gemm_wmma<float, 5, 1><<<grid, blk, 0, st>>>(g); break;
      default: gemm_wmma<float, 0, 1><<<grid, blk, 0, st>>>(g); break;
    }
  } else {
    switch (epi) {
      case 1:  gemm_wmma<float, 1, 0><<<grid, blk, 0, st>>>(g); break;
      case 2:  gemm_wmma<float, 2, 0><<<grid, blk, 0, st>>>(g); break;
      case 3:  gemm_wmma<float, 3, 0><<<grid, blk, 0, st>>>(g); break;
      case 4:  gemm_wmma<float, 4, 0><<<grid, blk, 0, st>>>(g); break;
      default: gemm_wmma<float, 0, 0><<<grid, blk, 0, st>>>(g); break;
    }
  }
}

// tri_mul params: ln_s,ln_b,Wa,ba,Wb,bb,Wg,bg,on_s,on_b,Wo,bo
static void tri_mul_host(hipStream_t st, void* const* d_in, int pb, float* z,
                         float* t0, float* t1, float* t2, float* t3, bf16_t* abf) {
  auto F = [&](int i) { return (const float*)d_in[i]; };
  const long LL = (long)B_ * L_ * L_;
  ln_kernel<<<dim3(LL / 8), dim3(256), 0, st>>>(z, (const float*)0, F(pb + 0), F(pb + 1), t0, LL, C_);
  run_gemm(st, 0, 0, GA(t0, F(pb + 2), F(pb + 3), nullptr, t1, (int)LL, C_, C_, C_, C_, C_, 0, 1.f));
  run_gemm(st, 3, 0, GA(t0, F(pb + 4), F(pb + 5), t1, t1, (int)LL, C_, C_, C_, C_, C_, 0, 1.f));
  run_gemm(st, 2, 0, GA(t0, F(pb + 6), F(pb + 7), nullptr, t2, (int)LL, C_, C_, C_, C_, C_, 0, 1.f));
  {
    dim3 g(L_ / 32, C_ / 32, B_ * L_), b(8, 32);
    transpose_to_bcik<<<g, b, 0, st>>>(t1, abf);   // a -> bf16 [b,c,i,k], convert once
  }
  // dominant FLOPs: 256 batched 256x256x256 bf16 WMMA GEMMs (A_c @ A_c^T)
  run_gemm(st, 0, 1, GA(abf, abf, nullptr, nullptr, t3, L_, L_, L_, L_, L_, L_, 1, 1.f,
                        B_ * C_, 1, (long)L_ * L_, 0, (long)L_ * L_, 0, (long)L_ * L_, 0));
  {
    dim3 g(L_ / 32, C_ / 32, B_ * L_), b(8, 32);
    transpose_to_bijc<<<g, b, 0, st>>>(t3, t0);
  }
  ln_kernel<<<dim3(LL / 8), dim3(256), 0, st>>>(t0, (const float*)0, F(pb + 8), F(pb + 9), t1, LL, C_);
  run_gemm(st, 4, 0, GA(t1, F(pb + 10), F(pb + 11), t2, z, (int)LL, C_, C_, C_, C_, C_, 0, 1.f));
}

// tri_att params: ln_s,ln_b,Wq,Wk,Wv,Wbias,Wg,bg,Wo,bo
static void tri_att_host(hipStream_t st, void* const* d_in, int pb, float* z,
                         float* t0, float* t1, float* t2, float* t3,
                         float* sbuf, float* attb) {
  auto F = [&](int i) { return (const float*)d_in[i]; };
  const long LL = (long)B_ * L_ * L_;
  const float inv_sqrt_c = 1.0f / sqrtf((float)C_);
  ln_kernel<<<dim3(LL / 8), dim3(256), 0, st>>>(z, (const float*)0, F(pb + 0), F(pb + 1), t0, LL, C_);
  run_gemm(st, 0, 0, GA(t0, F(pb + 2), nullptr, nullptr, t1, (int)LL, C_, C_, C_, C_, C_, 0, 1.f)); // q
  run_gemm(st, 0, 0, GA(t0, F(pb + 3), nullptr, nullptr, t2, (int)LL, C_, C_, C_, C_, C_, 0, 1.f)); // k
  rowdot_kernel<<<dim3((unsigned)(LL / 8)), dim3(256), 0, st>>>(t0, F(pb + 5), attb, LL);           // bias
  // s[b,i,j,k] = q@k^T / sqrt(C) + bias[b,i,j], batched over (b,i)
  run_gemm(st, 5, 0, GA(t1, t2, nullptr, attb, sbuf, L_, L_, C_, C_, C_, L_, 1, inv_sqrt_c,
                        B_ * L_, 1, (long)L_ * C_, 0, (long)L_ * C_, 0, (long)L_ * L_, 0, (long)L_, 0));
  softmax_dim2<<<dim3(B_ * L_), dim3(256), 0, st>>>(sbuf);  // softmax over j (dim=-2, faithful)
  run_gemm(st, 0, 0, GA(t0, F(pb + 4), nullptr, nullptr, t1, (int)LL, C_, C_, C_, C_, C_, 0, 1.f)); // v
  run_gemm(st, 2, 0, GA(t0, F(pb + 6), F(pb + 7), nullptr, t2, (int)LL, C_, C_, C_, C_, C_, 0, 1.f)); // g
  run_gemm(st, 4, 0, GA(sbuf, t1, nullptr, t2, t3, L_, C_, L_, L_, C_, C_, 0, 1.f,
                        B_ * L_, 1, (long)L_ * L_, 0, (long)L_ * C_, 0, (long)L_ * C_, 0, (long)L_ * C_, 0));
  run_gemm(st, 0, 0, GA(t3, F(pb + 8), F(pb + 9), nullptr, z, (int)LL, C_, C_, C_, C_, C_, 0, 1.f));
}

// ---------------------------------------------------------------------------
// kernel_launch: full forward pass
// Input order (setup_inputs dict order, recursive insertion order):
//  0 tokens | 1 emb | 2..17 enc0 | 18..33 enc1 | 34 exp_W | 35 exp_b
//  36..47 tm1 | 48..57 ta | 58..69 tm2 | 70..81 tm3 | 82 out_W | 83 out_b
// ---------------------------------------------------------------------------
extern "C" void kernel_launch(void* const* d_in, const int* in_sizes, int n_in,
                              void* d_out, int out_size, void* d_ws, size_t ws_size,
                              hipStream_t stream) {
  (void)in_sizes; (void)n_in; (void)out_size; (void)ws_size;
  auto F = [&](int i) { return (const float*)d_in[i]; };
  const int* tok = (const int*)d_in[0];

  const long BL = B_ * L_;                 // 512 token rows
  const long LL = (long)B_ * L_ * L_;      // 131072 pair rows

  char* ws = (char*)d_ws;
  size_t off = 0;
  auto alloc = [&](size_t bytes) -> char* {
    char* p = ws + off; off += (bytes + 255) & ~(size_t)255; return p;
  };
  float*  x      = (float*)alloc(BL * H_ * 4);
  float*  tq     = (float*)alloc(BL * H_ * 4);
  float*  tk     = (float*)alloc(BL * H_ * 4);
  float*  tv     = (float*)alloc(BL * H_ * 4);
  float*  tp     = (float*)alloc(BL * H_ * 4);
  float*  scores = (float*)alloc((size_t)B_ * NH_ * L_ * L_ * 4);
  float*  ffb    = (float*)alloc((size_t)BL * FF_ * 4);
  float*  pmat   = (float*)alloc((size_t)B_ * L_ * L_ * 4);
  float*  attb   = (float*)alloc((size_t)B_ * L_ * L_ * 4);
  float*  z      = (float*)alloc((size_t)LL * C_ * 4);
  float*  t0     = (float*)alloc((size_t)LL * C_ * 4);
  float*  t1     = (float*)alloc((size_t)LL * C_ * 4);
  float*  t2     = (float*)alloc((size_t)LL * C_ * 4);
  float*  t3     = (float*)alloc((size_t)LL * C_ * 4);
  bf16_t* abf    = (bf16_t*)alloc((size_t)LL * C_ * 2);
  float*  sbuf   = (float*)alloc((size_t)B_ * L_ * L_ * L_ * 4);

  // ---- embedding ----
  embed_kernel<<<dim3(BL), dim3(H_), 0, stream>>>(tok, F(1), x);

  // ---- 2 post-norm encoder layers ----
  for (int l = 0; l < 2; ++l) {
    int p = 2 + l * 16;  // Wq,bq,Wk,bk,Wv,bv,Wo,bo,ln1_s,ln1_b,W1,b1,W2,b2,ln2_s,ln2_b
    run_gemm(stream, 0, 0, GA(x, F(p + 0), F(p + 1), nullptr, tq, (int)BL, H_, H_, H_, H_, H_, 0, 1.f));
    run_gemm(stream, 0, 0, GA(x, F(p + 2), F(p + 3), nullptr, tk, (int)BL, H_, H_, H_, H_, H_, 0, 1.f));
    run_gemm(stream, 0, 0, GA(x, F(p + 4), F(p + 5), nullptr, tv, (int)BL, H_, H_, H_, H_, H_, 0, 1.f));
    run_gemm(stream, 0, 0, GA(tq, tk, nullptr, nullptr, scores, L_, L_, HD_, H_, H_, L_, 1, 0.125f,
                              B_, NH_, (long)L_ * H_, HD_, (long)L_ * H_, HD_,
                              (long)NH_ * L_ * L_, (long)L_ * L_));
    softmax_rows<<<dim3(B_ * NH_ * L_ / 8), dim3(256), 0, stream>>>(scores, (long)B_ * NH_ * L_, L_);
    run_gemm(stream, 0, 0, GA(scores, tv, nullptr, nullptr, tq, L_, HD_, L_, L_, H_, H_, 0, 1.f,
                              B_, NH_, (long)NH_ * L_ * L_, (long)L_ * L_,
                              (long)L_ * H_, HD_, (long)L_ * H_, HD_));
    run_gemm(stream, 0, 0, GA(tq, F(p + 6), F(p + 7), nullptr, tp, (int)BL, H_, H_, H_, H_, H_, 0, 1.f));
    ln_kernel<<<dim3(BL / 8), dim3(256), 0, stream>>>(x, tp, F(p + 8), F(p + 9), x, BL, H_);
    run_gemm(stream, 1, 0, GA(x, F(p + 10), F(p + 11), nullptr, ffb, (int)BL, FF_, H_, H_, FF_, FF_, 0, 1.f));
    run_gemm(stream, 0, 0, GA(ffb, F(p + 12), F(p + 13), nullptr, tp, (int)BL, H_, FF_, FF_, H_, H_, 0, 1.f));
    ln_kernel<<<dim3(BL / 8), dim3(256), 0, stream>>>(x, tp, F(p + 14), F(p + 15), x, BL, H_);
  }

  // ---- pairwise h @ h^T -> pmat; expand to [B,L,L,C] ----
  run_gemm(stream, 0, 0, GA(x, x, nullptr, nullptr, pmat, L_, L_, H_, H_, H_, L_, 1, 1.f,
                            B_, 1, (long)L_ * H_, 0, (long)L_ * H_, 0, (long)L_ * L_, 0));
  expand_kernel<<<dim3((unsigned)(LL / 4)), dim3(128), 0, stream>>>(pmat, F(34), F(35), z);

  // ---- triangle stack ----
  tri_mul_host(stream, d_in, 36, z, t0, t1, t2, t3, abf);        // tm1
  tri_att_host(stream, d_in, 48, z, t0, t1, t2, t3, sbuf, attb); // ta
  tri_mul_host(stream, d_in, 58, z, t0, t1, t2, t3, abf);        // tm2
  tri_mul_host(stream, d_in, 70, z, t0, t1, t2, t3, abf);        // tm3

  // ---- output head ----
  outproj_kernel<<<dim3((unsigned)(LL / 8)), dim3(256), 0, stream>>>(
      z, F(82), F(83), (float*)d_out, LL);
}
